// HandwritingRNN_70480413328197
// MI455X (gfx1250) — matrix operand order repaired and checked by reading, not compiled, not run on hardware
//
#include <hip/hip_runtime.h>

// ---------------------------------------------------------------------------
// Handwriting RNN (Graves) for MI455X / gfx1250, wave32, WMMA bf16.
//
//  - Weights packed once per launch to bf16 WMMA B-tile layout in ws (8.75MB,
//    L2-resident; HBM is touched once -> recurrence runs out of L2).
//  - Persistent kernel: 50 WGs x 256 threads (400 wave32 waves). Each gate
//    GEMM stage (64x1600, K in {480,896,1280}) is exactly 400 16x16 tiles ->
//    one tile per wave, K-loop of v_wmma_f32_16x16x32_bf16, f32 accumulate.
//  - Whole 64xKc activation matrix (<=160KB bf16, fits the 320KB WGP LDS) is
//    staged per stage with ONE bulk DMA: TENSOR_LOAD_TO_LDS (TDM, TENSORcnt)
//    for the LSTM stages, per-thread GLOBAL_LOAD_ASYNC_TO_LDS_B128 (ASYNCcnt)
//    for the GMM stage. K-loop then runs barrier-free from LDS.
//  - Device-wide sense-reversal barrier (atomics in ws + s_sleep): 8 stages
//    per timestep, 600 timesteps. Attention runs inside WG0 with local syncs.
// ---------------------------------------------------------------------------

typedef __attribute__((ext_vector_type(16))) __bf16 v16bf;
typedef __attribute__((ext_vector_type(8)))  float  v8f;
typedef __attribute__((ext_vector_type(4)))  unsigned v4u;
typedef __attribute__((ext_vector_type(8)))  int      v8i_;
typedef __attribute__((ext_vector_type(4)))  int      v4i_;

union ABfrag { v16bf v; uint4 q[2]; };

// ---- problem constants ----
#define LSTMN 400
#define MMIX  20
#define KATT  10
#define NALPH 77
#define BSZ   64
#define TLEN  600
#define TCHAR 50
#define NGATE 1600            // 4*LSTM

// padded combined-K per layer (multiple of 32)
#define KC1 480
#define KC2 896               // 880 padded
#define KC3 1280
#define KCG 1216              // 1200 padded
#define KT1 15
#define KT2 28
#define KT3 40
#define KTG 38
#define NT_GATE 100           // 1600/16
#define NT_GMM  8             // 128/16 (121 padded)

#define NWG      50
#define NTHREADS (NWG * 256)

constexpr unsigned LDS_BYTES = (unsigned)BSZ * KC3 * 2;  // 163,840 (<320KB WGP LDS)

// ---- workspace layout (bytes) ----
constexpr size_t OFF_SYNC = 0;                                   // 2 x u32
constexpr size_t OFF_W1   = 256;
constexpr size_t SZ_W1    = (size_t)NT_GATE * KT1 * 1024;
constexpr size_t OFF_W2   = OFF_W1 + SZ_W1;
constexpr size_t SZ_W2    = (size_t)NT_GATE * KT2 * 1024;
constexpr size_t OFF_W3   = OFF_W2 + SZ_W2;
constexpr size_t SZ_W3    = (size_t)NT_GATE * KT3 * 1024;
constexpr size_t OFF_WG   = OFF_W3 + SZ_W3;
constexpr size_t SZ_WG    = (size_t)NT_GMM * KTG * 1024;
constexpr size_t OFF_ACT1 = OFF_WG + SZ_WG;                      // bf16 64xKC1
constexpr size_t OFF_ACT2 = OFF_ACT1 + (size_t)BSZ * KC1 * 2;
constexpr size_t OFF_ACT3 = OFF_ACT2 + (size_t)BSZ * KC2 * 2;
constexpr size_t OFF_ACTG = OFF_ACT3 + (size_t)BSZ * KC3 * 2;
constexpr size_t OFF_GATE = OFF_ACTG + (size_t)BSZ * KCG * 2;    // f32 64x1600
constexpr size_t OFF_GOUT = OFF_GATE + (size_t)BSZ * NGATE * 4;  // f32 64x128
constexpr size_t OFF_H1   = OFF_GOUT + (size_t)BSZ * 128 * 4;    // f32 64x400
constexpr size_t OFF_C1   = OFF_H1 + (size_t)BSZ * LSTMN * 4;
constexpr size_t OFF_C2   = OFF_C1 + (size_t)BSZ * LSTMN * 4;
constexpr size_t OFF_C3   = OFF_C2 + (size_t)BSZ * LSTMN * 4;
constexpr size_t OFF_KAP  = OFF_C3 + (size_t)BSZ * LSTMN * 4;    // f32 64x16
constexpr size_t OFF_ABK  = OFF_KAP + (size_t)BSZ * 16 * 4;      // f32 64x32
constexpr size_t OFF_PHI  = OFF_ABK + (size_t)BSZ * 32 * 4;      // f32 64x64
constexpr size_t OFF_BS   = OFF_PHI + (size_t)BSZ * 64 * 4;      // f32 3x1600
constexpr size_t OFF_END  = OFF_BS + (size_t)3 * NGATE * 4;

constexpr size_t ZERO_BEGIN = OFF_ACT1;
constexpr size_t ZERO_BYTES = OFF_END - OFF_ACT1;
constexpr size_t ZERO_N16   = ZERO_BYTES / 16;

// ---------------------------------------------------------------------------
__device__ __forceinline__ float sigf(float x) { return 1.0f / (1.0f + expf(-x)); }

__device__ __forceinline__ void grid_barrier(unsigned* cnt, unsigned* gen) {
  __syncthreads();
  if (threadIdx.x == 0) {
    unsigned old = __hip_atomic_load(gen, __ATOMIC_RELAXED, __HIP_MEMORY_SCOPE_AGENT);
    unsigned arr = __hip_atomic_fetch_add(cnt, 1u, __ATOMIC_ACQ_REL, __HIP_MEMORY_SCOPE_AGENT);
    if (arr == NWG - 1) {
      __hip_atomic_store(cnt, 0u, __ATOMIC_RELAXED, __HIP_MEMORY_SCOPE_AGENT);
      __hip_atomic_fetch_add(gen, 1u, __ATOMIC_ACQ_REL, __HIP_MEMORY_SCOPE_AGENT);
    } else {
      while (__hip_atomic_load(gen, __ATOMIC_ACQUIRE, __HIP_MEMORY_SCOPE_AGENT) == old)
        __builtin_amdgcn_s_sleep(1);
    }
  }
  __syncthreads();
}

// Per-lane 16B async DMA global -> LDS (ASYNCcnt).
__device__ __forceinline__ void async_lds_b128(unsigned ldsOff, const void* gsrc) {
  asm volatile("global_load_async_to_lds_b128 %0, %1, off"
               :: "v"(ldsOff), "v"(gsrc) : "memory");
}
__device__ __forceinline__ void wait_async0() {
  asm volatile("s_wait_asynccnt 0x0" ::: "memory");
}

// Bulk-stage the full 64xKc bf16 activation matrix into LDS with one TDM
// descriptor (D# groups per CDNA5 ISA):
//   2D tensor: dim0 = Kc elems (data_size=2B), dim1 = 64 rows, stride0 = Kc;
//   tile = whole tensor. Issued by wave 0 only; TENSORcnt wait; WG barrier.
__device__ __forceinline__ void stage_act_tdm(const __bf16* act, unsigned Kc,
                                              __bf16* sact, int tid) {
  if (tid < 32) {  // one TDM op issued by wave 0 (EXEC-ignored instruction)
    const unsigned long long ga = (unsigned long long)(const void*)act;
    const unsigned ldsAddr = (unsigned)(size_t)(void*)sact;
    v4u g0 = { 1u,                                  // count=1 (valid D#)
               ldsAddr,                             // lds_addr [63:32]
               (unsigned)ga,                        // global_addr [95:64]
               (unsigned)((ga >> 32) & 0x01ffffffu) | 0x80000000u }; // type=2
    v8i_ g1 = { (int)(1u << 16),                    // data_size=1 (2 bytes)
                (int)((Kc & 0xffffu) << 16),        // tensor_dim0 lo
                (int)((Kc >> 16) | (64u << 16)),    // tensor_dim0 hi | dim1 lo
                (int)(Kc << 16),                    // dim1 hi=0 | tile_dim0=Kc
                (int)64,                            // tile_dim1=64 | tile_dim2=0
                (int)Kc,                            // tensor_dim0_stride lo
                0, 0 };
    v4i_ gz = { 0, 0, 0, 0 };
#if defined(__clang_major__) && (__clang_major__ >= 23)
    v8i_ gz8 = { 0, 0, 0, 0, 0, 0, 0, 0 };
    __builtin_amdgcn_tensor_load_to_lds(g0, g1, gz, gz, gz8, 0);
#else
    __builtin_amdgcn_tensor_load_to_lds(g0, g1, gz, gz, 0);
#endif
    __builtin_amdgcn_s_wait_tensorcnt(0);
  }
  __syncthreads();
}

// Bulk-stage via per-thread async b128 DMA (contiguous copy), ASYNCcnt path.
__device__ __forceinline__ void stage_act_async(const __bf16* act, int Kc,
                                                __bf16* sact, int tid) {
  const unsigned ldsBase = (unsigned)(size_t)(void*)sact;
  const char* src = (const char*)act;
  const int chunks = (BSZ * Kc * 2) / 16;
  for (int c = tid; c < chunks; c += 256)
    async_lds_b128(ldsBase + (unsigned)c * 16u, src + (size_t)c * 16);
  wait_async0();
  __syncthreads();
}

// One 16x16 output tile per wave; whole K-loop runs barrier-free out of LDS.
__device__ __forceinline__ void gate_tile_lds(
    int Kc, int Kt, const unsigned char* __restrict__ wpack,
    float* __restrict__ gout, int goutStride,
    int tile, int lane, const __bf16* sact)
{
  const int mt = tile & 3, nt = tile >> 2;
  const int lo = lane & 15, hi = lane >> 4;
  const int base = hi * 16;
  const int rr = mt * 16 + lo;                    // A row this lane owns
  const __bf16* arow = sact + (size_t)rr * Kc;
  v8f acc = {};
  for (int kt = 0; kt < Kt; ++kt) {
    ABfrag a, b;
    a.q[0] = *(const uint4*)(arow + kt * 32 + base);
    a.q[1] = *(const uint4*)(arow + kt * 32 + base + 8);
    const unsigned char* wt = wpack + ((size_t)nt * Kt + kt) * 1024 + lane * 32;
    b.q[0] = *(const uint4*)(wt);
    b.q[1] = *(const uint4*)(wt + 16);
    if (kt + 1 < Kt)
      __builtin_prefetch(wpack + ((size_t)nt * Kt + kt + 1) * 1024 + lane * 32, 0, 1);
    acc = __builtin_amdgcn_wmma_f32_16x16x32_bf16(
        false, a.v, false, b.v, (short)0, acc, false, false);
  }
  // C layout: VGPR v, lanes0-15 -> M=v, lanes16-31 -> M=v+8; N = lane&15
  float* gb = gout + (size_t)(mt * 16 + hi * 8) * goutStride + nt * 16 + lo;
#pragma unroll
  for (int v = 0; v < 8; ++v) gb[(size_t)v * goutStride] = acc[v];
}

__device__ __forceinline__ void lstm_elem(
    const float* __restrict__ gate, const float* __restrict__ bs,
    float* __restrict__ c, float* __restrict__ hf,
    __bf16* s0, int st0, int o0, __bf16* s1, int st1, int o1,
    __bf16* s2, int st2, int o2, __bf16* s3, int st3, int o3, int gtid)
{
  for (int idx = gtid; idx < BSZ * LSTMN; idx += NTHREADS) {
    const int b = idx / LSTMN, u = idx - b * LSTMN;
    const float* g = gate + (size_t)b * NGATE;
    const float gi = g[u]             + bs[u];
    const float gf = g[LSTMN + u]     + bs[LSTMN + u];
    const float gg = g[2 * LSTMN + u] + bs[2 * LSTMN + u];
    const float go = g[3 * LSTMN + u] + bs[3 * LSTMN + u];
    const float cn = sigf(gf) * c[idx] + sigf(gi) * tanhf(gg);
    c[idx] = cn;
    const float h = sigf(go) * tanhf(cn);
    if (hf) hf[idx] = h;
    const __bf16 hb = (__bf16)h;
    if (s0) s0[(size_t)b * st0 + o0 + u] = hb;
    if (s1) s1[(size_t)b * st1 + o1 + u] = hb;
    if (s2) s2[(size_t)b * st2 + o2 + u] = hb;
    if (s3) s3[(size_t)b * st3 + o3 + u] = hb;
  }
}

// ---------------------------------------------------------------------------
__global__ void hw455_zero(unsigned char* ws) {
  size_t i = (size_t)blockIdx.x * blockDim.x + threadIdx.x;
  if (i < ZERO_N16) {
    uint4 z; z.x = z.y = z.z = z.w = 0u;
    ((uint4*)(ws + ZERO_BEGIN))[i] = z;
  }
  if (blockIdx.x == 0 && threadIdx.x == 0) {
    ((unsigned*)(ws + OFF_SYNC))[0] = 0u;
    ((unsigned*)(ws + OFF_SYNC))[1] = 0u;
  }
}

__global__ void hw455_setup(
    const float* __restrict__ bih1, const float* __restrict__ bhh1,
    const float* __restrict__ bih2, const float* __restrict__ bhh2,
    const float* __restrict__ bih3, const float* __restrict__ bhh3,
    const float* __restrict__ inputs, unsigned char* ws)
{
  int idx = blockIdx.x * 256 + threadIdx.x;
  if (idx < 3 * NGATE) {
    int l = idx / NGATE, n = idx - l * NGATE;
    const float* a = (l == 0) ? bih1 : (l == 1) ? bih2 : bih3;
    const float* b = (l == 0) ? bhh1 : (l == 1) ? bhh2 : bhh3;
    ((float*)(ws + OFF_BS))[idx] = a[n] + b[n];
  } else if (idx < 3 * NGATE + BSZ * 3) {
    int j = idx - 3 * NGATE, b = j / 3, cch = j - b * 3;
    __bf16 xv = (__bf16)inputs[(size_t)b * TLEN * 3 + cch];  // t = 0
    ((__bf16*)(ws + OFF_ACT1))[(size_t)b * KC1 + 77 + cch] = xv;
    ((__bf16*)(ws + OFF_ACT2))[(size_t)b * KC2 + cch]      = xv;
    ((__bf16*)(ws + OFF_ACT3))[(size_t)b * KC3 + cch]      = xv;
  }
}

// Pack fp32 weights -> bf16 WMMA-B tiles (1KB per 32x16 tile, lane-major:
// lane L owns column N = L%16, K half = (L/16)*16, 16 sequential K values).
__global__ void hw455_pack(
    const float* __restrict__ Wih1, const float* __restrict__ Whh1,
    const float* __restrict__ Wih2, const float* __restrict__ Whh2,
    const float* __restrict__ Wih3, const float* __restrict__ Whh3,
    const float* __restrict__ Wgmm, unsigned char* ws)
{
  constexpr int S1 = NT_GATE * KT1 * 32;
  constexpr int S2 = NT_GATE * KT2 * 32;
  constexpr int S3 = NT_GATE * KT3 * 32;
  constexpr int SG = NT_GMM  * KTG * 32;
  int idx = blockIdx.x * 256 + threadIdx.x;
  int layer, rel;
  if      (idx < S1)                { layer = 0; rel = idx; }
  else if (idx < S1 + S2)           { layer = 1; rel = idx - S1; }
  else if (idx < S1 + S2 + S3)      { layer = 2; rel = idx - S1 - S2; }
  else if (idx < S1 + S2 + S3 + SG) { layer = 3; rel = idx - S1 - S2 - S3; }
  else return;
  const int tile = rel >> 5, lane = rel & 31;
  const int Kt = (layer == 0) ? KT1 : (layer == 1) ? KT2 : (layer == 2) ? KT3 : KTG;
  const int kt = tile % Kt, nt = tile / Kt;
  const size_t off = (layer == 0) ? OFF_W1 : (layer == 1) ? OFF_W2
                   : (layer == 2) ? OFF_W3 : OFF_WG;
  __bf16* dst = (__bf16*)(ws + off + (size_t)tile * 1024 + lane * 32);
  const int lo = lane & 15, khalf = (lane >> 4) * 16;
  const int n = nt * 16 + lo;
#pragma unroll 4
  for (int j = 0; j < 16; ++j) {
    const int k = kt * 32 + khalf + j;
    float v = 0.0f;
    if (layer == 0)      v = (k < 80)  ? Wih1[(size_t)n * 80  + k] : Whh1[(size_t)n * 400 + (k - 80)];
    else if (layer == 1) v = (k < 480) ? Wih2[(size_t)n * 480 + k]
                           : (k < 880) ? Whh2[(size_t)n * 400 + (k - 480)] : 0.0f;
    else if (layer == 2) v = (k < 880) ? Wih3[(size_t)n * 880 + k] : Whh3[(size_t)n * 400 + (k - 880)];
    else                 v = (n < 121 && k < 1200) ? Wgmm[(size_t)n * 1200 + k] : 0.0f;
    dst[j] = (__bf16)v;
  }
}

// ---------------------------------------------------------------------------
__global__ __launch_bounds__(256) void hw455_main(
    const float* __restrict__ inputs, const int* __restrict__ cs,
    const int* __restrict__ cslen, const float* __restrict__ bias,
    const float* __restrict__ Watt, const float* __restrict__ batt,
    const float* __restrict__ bgmm, float* __restrict__ out,
    unsigned char* __restrict__ ws)
{
  const int tid  = threadIdx.x;
  const int gtid = blockIdx.x * 256 + tid;
  const int gw   = gtid >> 5;
  const int lane = tid & 31;
  const int wg   = blockIdx.x;
  extern __shared__ __bf16 sact[];       // 160KB dynamic LDS (full act matrix)

  unsigned* cnt = (unsigned*)(ws + OFF_SYNC);
  unsigned* gen = cnt + 1;
  __bf16* act1 = (__bf16*)(ws + OFF_ACT1);
  __bf16* act2 = (__bf16*)(ws + OFF_ACT2);
  __bf16* act3 = (__bf16*)(ws + OFF_ACT3);
  __bf16* actG = (__bf16*)(ws + OFF_ACTG);
  float* gate = (float*)(ws + OFF_GATE);
  float* gout = (float*)(ws + OFF_GOUT);
  float* h1f  = (float*)(ws + OFF_H1);
  float* c1   = (float*)(ws + OFF_C1);
  float* c2   = (float*)(ws + OFF_C2);
  float* c3   = (float*)(ws + OFF_C3);
  float* kap_ = (float*)(ws + OFF_KAP);
  float* abk  = (float*)(ws + OFF_ABK);
  float* phi  = (float*)(ws + OFF_PHI);
  const float* bs1 = (const float*)(ws + OFF_BS);
  const float* bs2 = bs1 + NGATE;
  const float* bs3 = bs2 + NGATE;

  for (int t = 0; t < TLEN; ++t) {
    // ---- LSTM1 gates: [win,x,h1] (K=480) @ W1 ----
    stage_act_tdm(act1, KC1, sact, tid);
    gate_tile_lds(KC1, KT1, ws + OFF_W1, gate, NGATE, gw, lane, sact);
    grid_barrier(cnt, gen);
    lstm_elem(gate, bs1, c1, h1f,
              act1, KC1, 80, act2, KC2, 3, act3, KC3, 3, actG, KCG, 0, gtid);
    grid_barrier(cnt, gen);
    // ---- attention (entirely inside WG0, local syncs) ----
    if (wg == 0) {
      // abk = exp(h1 @ Watt^T + b_att): 64x30 dots of K=400, float4
      for (int i = tid; i < BSZ * 30; i += 256) {
        const int b = i / 30, j = i - b * 30;
        const float4* hr = (const float4*)(h1f + (size_t)b * LSTMN);
        const float4* wr = (const float4*)(Watt + (size_t)j * LSTMN);
        float d = 0.0f;
        for (int k = 0; k < LSTMN / 4; ++k) {
          const float4 hv = hr[k], wv = wr[k];
          d += hv.x * wv.x + hv.y * wv.y + hv.z * wv.z + hv.w * wv.w;
        }
        abk[b * 32 + j] = expf(d + batt[j]);
      }
      __syncthreads();
      // kappa += kstep; phi over char positions (masked)
      if (tid < BSZ) {
        const int b = tid;
        float kap[KATT];
#pragma unroll
        for (int k = 0; k < KATT; ++k) {
          float nk = kap_[b * 16 + k] + abk[b * 32 + 20 + k];
          kap_[b * 16 + k] = nk;
          kap[k] = nk;
        }
        const int len = cslen[b];
        for (int u = 0; u < TCHAR; ++u) {
          float p = 0.0f;
          if (u < len) {
#pragma unroll
            for (int k = 0; k < KATT; ++k) {
              float dd = kap[k] - (float)u;
              p += abk[b * 32 + k] * expf(-abk[b * 32 + 10 + k] * dd * dd);
            }
          }
          phi[b * 64 + u] = p;
        }
      }
      __syncthreads();
      // window = phi @ one_hot(chars) -> act1/act2/act3 window slots
      for (int i = tid; i < BSZ * NALPH; i += 256) {
        const int b = i / NALPH, a = i - b * NALPH;
        const int* crow = cs + (size_t)b * TCHAR;
        const float* pr = phi + b * 64;
        float w = 0.0f;
        for (int u = 0; u < TCHAR; ++u) if (crow[u] == a) w += pr[u];
        const __bf16 wb = (__bf16)w;
        act1[(size_t)b * KC1 + a]       = wb;
        act2[(size_t)b * KC2 + 403 + a] = wb;
        act3[(size_t)b * KC3 + 803 + a] = wb;
      }
      // stage x_{t+1} for next step's LSTM1 (act1 only; act2/3 still need x_t)
      if (t + 1 < TLEN) {
        for (int i = tid; i < BSZ * 3; i += 256) {
          const int b = i / 3, cch = i - b * 3;
          act1[(size_t)b * KC1 + 77 + cch] =
              (__bf16)inputs[((size_t)b * TLEN + (t + 1)) * 3 + cch];
        }
      }
    }
    grid_barrier(cnt, gen);
    // ---- LSTM2 gates (K=896 incl pad) ----
    stage_act_tdm(act2, KC2, sact, tid);
    gate_tile_lds(KC2, KT2, ws + OFF_W2, gate, NGATE, gw, lane, sact);
    grid_barrier(cnt, gen);
    lstm_elem(gate, bs2, c2, nullptr,
              act2, KC2, 480, act3, KC3, 403, actG, KCG, 400,
              nullptr, 0, 0, gtid);
    grid_barrier(cnt, gen);
    // ---- LSTM3 gates (K=1280) ----
    stage_act_tdm(act3, KC3, sact, tid);
    gate_tile_lds(KC3, KT3, ws + OFF_W3, gate, NGATE, gw, lane, sact);
    grid_barrier(cnt, gen);
    lstm_elem(gate, bs3, c3, nullptr,
              act3, KC3, 880, actG, KCG, 800, nullptr, 0, 0,
              nullptr, 0, 0, gtid);
    grid_barrier(cnt, gen);
    // ---- GMM head: [h1,h2,h3] (K=1216 incl pad) -> 121 (pad 128) ----
    if (wg < 4) {
      stage_act_async(actG, KCG, sact, tid);
      gate_tile_lds(KCG, KTG, ws + OFF_WG, gout, 128, gw, lane, sact);
    }
    grid_barrier(cnt, gen);
    // ---- GMM post-process -> y; plus stage x_{t+1} for LSTM2/3 ----
    if (gtid < BSZ) {
      const int b = gtid;
      const float* r = gout + (size_t)b * 128;
      const float bb = bias[b];
      float* y = out + ((size_t)b * TLEN + t) * 121;
      float ph[MMIX];
      float mx = -1e30f;
#pragma unroll
      for (int m = 0; m < MMIX; ++m) {
        ph[m] = (r[1 + m] + bgmm[1 + m]) * (1.0f + bb);
        mx = fmaxf(mx, ph[m]);
      }
      float s = 0.0f;
#pragma unroll
      for (int m = 0; m < MMIX; ++m) { float e = expf(ph[m] - mx); ph[m] = e; s += e; }
      const float inv = 1.0f / s;
#pragma unroll
      for (int m = 0; m < MMIX; ++m) y[m] = ph[m] * inv;
      for (int j = 0; j < 2 * MMIX; ++j)
        y[20 + j] = expf(r[61 + j] + bgmm[61 + j] - bb);
      for (int j = 0; j < MMIX; ++j)
        y[60 + j] = tanhf(r[101 + j] + bgmm[101 + j]);
      for (int j = 0; j < 2 * MMIX; ++j)
        y[80 + j] = r[21 + j] + bgmm[21 + j];
      y[120] = 1.0f / (1.0f + expf(-(r[0] + bgmm[0])));
    } else if (gtid < BSZ + BSZ * 3 && t + 1 < TLEN) {
      const int j = gtid - BSZ, b = j / 3, cch = j - b * 3;
      const __bf16 xv = (__bf16)inputs[((size_t)b * TLEN + (t + 1)) * 3 + cch];
      act2[(size_t)b * KC2 + cch] = xv;
      act3[(size_t)b * KC3 + cch] = xv;
    }
    // no trailing barrier: next step's first writes sit behind barrier #1.
  }
}

// ---------------------------------------------------------------------------
extern "C" void kernel_launch(void* const* d_in, const int* in_sizes, int n_in,
                              void* d_out, int out_size, void* d_ws, size_t ws_size,
                              hipStream_t stream) {
  (void)in_sizes; (void)n_in; (void)out_size; (void)ws_size;  // need ~10.3 MB ws
  const float* inputs = (const float*)d_in[0];
  const int*   cs     = (const int*)  d_in[1];
  const int*   cslen  = (const int*)  d_in[2];
  const float* bias   = (const float*)d_in[3];
  const float* Wih1   = (const float*)d_in[4];
  const float* Whh1   = (const float*)d_in[5];
  const float* bih1   = (const float*)d_in[6];
  const float* bhh1   = (const float*)d_in[7];
  const float* Wih2   = (const float*)d_in[8];
  const float* Whh2   = (const float*)d_in[9];
  const float* bih2   = (const float*)d_in[10];
  const float* bhh2   = (const float*)d_in[11];
  const float* Wih3   = (const float*)d_in[12];
  const float* Whh3   = (const float*)d_in[13];
  const float* bih3   = (const float*)d_in[14];
  const float* bhh3   = (const float*)d_in[15];
  const float* Watt   = (const float*)d_in[16];
  const float* batt   = (const float*)d_in[17];
  const float* Wgmm   = (const float*)d_in[18];
  const float* bgmm   = (const float*)d_in[19];
  float* out = (float*)d_out;
  unsigned char* ws = (unsigned char*)d_ws;

  const int zeroBlocks = (int)((ZERO_N16 + 255) / 256);
  hw455_zero<<<zeroBlocks, 256, 0, stream>>>(ws);
  hw455_setup<<<(3 * NGATE + BSZ * 3 + 255) / 256, 256, 0, stream>>>(
      bih1, bhh1, bih2, bhh2, bih3, bhh3, inputs, ws);
  const int packThreads = NT_GATE * (KT1 + KT2 + KT3) * 32 + NT_GMM * KTG * 32;
  hw455_pack<<<(packThreads + 255) / 256, 256, 0, stream>>>(
      Wih1, Whh1, Wih2, Whh2, Wih3, Whh3, Wgmm, ws);
  hw455_main<<<NWG, 256, LDS_BYTES, stream>>>(
      inputs, cs, cslen, bias, Watt, batt, bgmm, out, ws);
}
